// mlp_full_forward_model_21689584845243
// MI455X (gfx1250) — matrix operand order repaired and checked by reading, not compiled, not run on hardware
//
#include <hip/hip_runtime.h>
#include <stdint.h>

// ---------------------------------------------------------------------------
// CDNA5 (gfx1250) fused GraphNet forward using v_wmma_f32_16x16x32_f16.
// wave32; 16x16x32 WMMA tiles; A staged in LDS (f16), B pre-packed in ws (f16).
// Round 3: fix async global->LDS builtin pointer types (AS1 v4i* / AS3 v4i*).
// ---------------------------------------------------------------------------

typedef __attribute__((ext_vector_type(16))) _Float16 v16h;
typedef __attribute__((ext_vector_type(8)))  _Float16 v8h;
typedef __attribute__((ext_vector_type(8)))  float    v8f;
typedef __attribute__((ext_vector_type(4)))  int      v4i;

union F16Frag { v16h v; v8h h[2]; };

#if defined(__has_builtin)
#if __has_builtin(__builtin_amdgcn_global_load_async_to_lds_b128) && \
    __has_builtin(__builtin_amdgcn_s_wait_asynccnt)
#define GNN_ASYNC_LDS 1
#endif
#endif

#ifdef GNN_ASYNC_LDS
typedef __attribute__((address_space(1))) v4i* gv4i_p;  // global
typedef __attribute__((address_space(3))) v4i* lv4i_p;  // LDS
#endif

// A fragment (16x32 f16, M x K) from row-major f16 LDS buffer.
// Lane l<16 : row M=l,    K = {k0..k0+7, k0+16..k0+23}
// Lane l>=16: row M=l-16, K = {k0+8..k0+15, k0+24..k0+31}
__device__ __forceinline__ v16h lds_a_frag(const _Float16* base, int rowBase,
                                           int stride, int k0, int lane) {
  const int half = lane >> 4;
  const int l    = lane & 15;
  const _Float16* p = base + (size_t)(rowBase + l) * stride + k0 + half * 8;
  F16Frag u;
  u.h[0] = *(const v8h*)(p);
  u.h[1] = *(const v8h*)(p + 16);
  return u.v;
}

// B fragment (32x16 f16, K x N) from pre-packed global buffer:
// per (frag, lane) 16 contiguous f16 = column n, K = {8h..8h+7, 16+8h..16+8h+7}.
__device__ __forceinline__ v16h glb_b_frag(const _Float16* pack, int frag, int lane) {
  const v8h* p = (const v8h*)(pack + ((size_t)frag * 32 + lane) * 16);
  F16Frag u;
  u.h[0] = p[0];
  u.h[1] = p[1];
  return u.v;
}

__device__ __forceinline__ v8f wmma_f16(v16h a, v16h b, v8f c) {
  return __builtin_amdgcn_wmma_f32_16x16x32_f16(false, a, false, b, (short)0, c,
                                                false, false);
}

// Convert 8 contiguous f32 -> v8h (two b128 loads + packed converts).
__device__ __forceinline__ v8h cvt8(const float* __restrict__ s) {
  float4 a = *(const float4*)s;
  float4 b = *(const float4*)(s + 4);
  v8h r = {(_Float16)a.x, (_Float16)a.y, (_Float16)a.z, (_Float16)a.w,
           (_Float16)b.x, (_Float16)b.y, (_Float16)b.z, (_Float16)b.w};
  return r;
}

// 16B copy global f16 -> LDS f16. Uses CDNA5 async global->LDS (ASYNCcnt).
__device__ __forceinline__ void cp16(_Float16* dst_lds,
                                     const _Float16* __restrict__ src) {
#ifdef GNN_ASYNC_LDS
  __builtin_amdgcn_global_load_async_to_lds_b128(
      (gv4i_p)(uintptr_t)src,
      (lv4i_p)(unsigned)(uintptr_t)dst_lds, 0, 0);
#else
  *(v8h*)dst_lds = *(const v8h*)src;
#endif
}

__device__ __forceinline__ void async_fence() {
#ifdef GNN_ASYNC_LDS
  __builtin_amdgcn_s_wait_asynccnt(0);
#endif
}

// GEMM1: H = relu(A @ W1 + b1), A is 16 x (KT*32) in LDS (stride STRIDE),
// result 16 x (NT*16) stored as f16 into Hs (stride 128).
template <int KT, int NT, int STRIDE>
__device__ __forceinline__ void gemm1_relu(const _Float16* Ain, int rowBase,
                                           const _Float16* wp,
                                           const float* __restrict__ bias,
                                           _Float16* Hs, int lane) {
  const int half  = lane >> 4;
  const int nlane = lane & 15;
  v8f acc[NT];
#pragma unroll
  for (int nt = 0; nt < NT; ++nt) {
    float bb = bias[nt * 16 + nlane];
    v8f t = {bb, bb, bb, bb, bb, bb, bb, bb};
    acc[nt] = t;
  }
  constexpr int BLK = (NT >= 4) ? 4 : NT;
#pragma unroll
  for (int kt = 0; kt < KT; ++kt) {
    v16h a = lds_a_frag(Ain, rowBase, STRIDE, kt * 32, lane);
#pragma unroll
    for (int nb = 0; nb < NT; nb += BLK) {
      v16h bf[BLK];
#pragma unroll
      for (int j = 0; j < BLK; ++j)
        bf[j] = glb_b_frag(wp, kt * NT + nb + j, lane);
#pragma unroll
      for (int j = 0; j < BLK; ++j)
        acc[nb + j] = wmma_f16(a, bf[j], acc[nb + j]);
    }
  }
  const int hrow = rowBase + half * 8;
#pragma unroll
  for (int nt = 0; nt < NT; ++nt) {
    int hc = nt * 16 + nlane;
#pragma unroll
    for (int g = 0; g < 8; ++g) {
      float v = acc[nt][g];
      Hs[(size_t)(hrow + g) * 128 + hc] = (_Float16)(v > 0.f ? v : 0.f);
    }
  }
}

// GEMM2: acc = H @ W2 + b2 ; H is 16 x 128 f16 in LDS.
template <int NTO>
__device__ __forceinline__ void gemm2(const _Float16* Hs, int rowBase,
                                      const _Float16* wp,
                                      const float* __restrict__ bias, int nreal,
                                      int lane, v8f acc[NTO]) {
  const int nlane = lane & 15;
#pragma unroll
  for (int nt = 0; nt < NTO; ++nt) {
    int n = nt * 16 + nlane;
    float bb = (n < nreal) ? bias[n] : 0.f;
    v8f t = {bb, bb, bb, bb, bb, bb, bb, bb};
    acc[nt] = t;
  }
  constexpr int BLK = (NTO >= 4) ? 4 : NTO;
#pragma unroll
  for (int kt = 0; kt < 4; ++kt) {  // K = 128 for all second layers
    v16h a = lds_a_frag(Hs, rowBase, 128, kt * 32, lane);
#pragma unroll
    for (int nb = 0; nb < NTO; nb += BLK) {
      v16h bf[BLK];
#pragma unroll
      for (int j = 0; j < BLK; ++j)
        bf[j] = glb_b_frag(wp, kt * NTO + nb + j, lane);
#pragma unroll
      for (int j = 0; j < BLK; ++j)
        acc[nb + j] = wmma_f16(a, bf[j], acc[nb + j]);
    }
  }
}

// ---------------------------------------------------------------------------
// Weight repack: f32 row-major W[K][N] -> f16 WMMA B fragments (zero padded).
// ---------------------------------------------------------------------------
__global__ void k_repack(const float* __restrict__ W, _Float16* __restrict__ out,
                         int K, int N, int NT, int nfrag) {
  int idx  = blockIdx.x * blockDim.x + threadIdx.x;
  int frag = idx >> 5;
  int lane = idx & 31;
  if (frag >= nfrag) return;
  int kt    = frag / NT;
  int nt    = frag % NT;
  int kbase = kt * 32 + ((lane >> 4) ? 8 : 0);
  int n     = nt * 16 + (lane & 15);
  _Float16* o = out + ((size_t)frag * 32 + lane) * 16;
#pragma unroll
  for (int j = 0; j < 16; ++j) {
    int k   = kbase + (j < 8 ? j : j + 8);
    float v = (k < K && n < N) ? W[(size_t)k * N + n] : 0.f;
    o[j] = (_Float16)v;
  }
}

__global__ void k_zero(float* __restrict__ p, long long n) {
  long long i = (long long)blockIdx.x * blockDim.x + threadIdx.x;
  if (i < n) p[i] = 0.f;
}

// ---------------------------------------------------------------------------
// GN1 edge MLP: in 48 (pad 64) -> 128 relu -> 64 ; store e1 f16 + atomic agg1
// ---------------------------------------------------------------------------
__global__ __launch_bounds__(128) void k_edge1(
    const float* __restrict__ x, const float* __restrict__ eattr,
    const int* __restrict__ row, const int* __restrict__ col,
    const _Float16* __restrict__ w1p, const float* __restrict__ b1,
    const _Float16* __restrict__ w2p, const float* __restrict__ b2,
    _Float16* __restrict__ e1ws, float* __restrict__ agg1, int E) {
  __shared__ alignas(16) _Float16 Ain[64 * 64];
  __shared__ alignas(16) _Float16 Hs[64 * 128];
  __shared__ int rowL[64], colL[64];
  const int tid   = threadIdx.x;
  const int ebase = blockIdx.x * 64;
  if (tid < 64) {
    int e = ebase + tid;
    rowL[tid] = (e < E) ? row[e] : 0;
    colL[tid] = (e < E) ? col[e] : 0;
  }
  __syncthreads();
  // Stage 64 rows x 64 f16, 16B chunks (8 per row).
  for (int t = tid; t < 64 * 8; t += 128) {
    int r = t >> 3, c = t & 7;
    int e = ebase + r;
    int ec = (e < E) ? e : (E - 1);
    v8h hv;
    if (c < 2)      hv = cvt8(x + (size_t)rowL[r] * 16 + c * 8);
    else if (c < 4) hv = cvt8(x + (size_t)colL[r] * 16 + (c - 2) * 8);
    else if (c < 6) hv = cvt8(eattr + (size_t)ec * 16 + (c - 4) * 8);
    else            hv = (v8h){};
    *(v8h*)&Ain[r * 64 + c * 8] = hv;
  }
  __syncthreads();
  const int lane = tid & 31, wave = tid >> 5;
  const int rowBase = wave * 16;
  gemm1_relu<2, 8, 64>(Ain, rowBase, w1p, b1, Hs, lane);
  __syncthreads();
  v8f acc[4];
  gemm2<4>(Hs, rowBase, w2p, b2, 64, lane, acc);
  const int half = lane >> 4, nlane = lane & 15;
#pragma unroll
  for (int nt = 0; nt < 4; ++nt) {
    int cn = nt * 16 + nlane;
#pragma unroll
    for (int g = 0; g < 8; ++g) {
      int r = rowBase + half * 8 + g;
      int e = ebase + r;
      if (e < E) {
        float v = acc[nt][g];
        e1ws[(size_t)e * 64 + cn] = (_Float16)v;
        atomicAdd(&agg1[(size_t)colL[r] * 64 + cn], v);
      }
    }
  }
}

// ---------------------------------------------------------------------------
// GN1 node MLP: in 80 (pad 96) -> 128 relu -> 128 ; store x1 f16
// ---------------------------------------------------------------------------
__global__ __launch_bounds__(128) void k_node1(
    const float* __restrict__ x, const float* __restrict__ agg1,
    const _Float16* __restrict__ w1p, const float* __restrict__ b1,
    const _Float16* __restrict__ w2p, const float* __restrict__ b2,
    _Float16* __restrict__ x1ws, int N) {
  __shared__ alignas(16) _Float16 Ain[64 * 96];
  __shared__ alignas(16) _Float16 Hs[64 * 128];
  const int tid   = threadIdx.x;
  const int nbase = blockIdx.x * 64;
  // 12 chunks/row: 2 from x, 8 from agg1, 2 zero pad.
  for (int t = tid; t < 64 * 12; t += 128) {
    int r = t / 12, c = t % 12;
    int nd = nbase + r;
    int nc = (nd < N) ? nd : (N - 1);
    v8h hv;
    if (c < 2)       hv = cvt8(x + (size_t)nc * 16 + c * 8);
    else if (c < 10) hv = cvt8(agg1 + (size_t)nc * 64 + (c - 2) * 8);
    else             hv = (v8h){};
    *(v8h*)&Ain[r * 96 + c * 8] = hv;
  }
  __syncthreads();
  const int lane = tid & 31, wave = tid >> 5;
  const int rowBase = wave * 16;
  gemm1_relu<3, 8, 96>(Ain, rowBase, w1p, b1, Hs, lane);
  __syncthreads();
  v8f acc[8];
  gemm2<8>(Hs, rowBase, w2p, b2, 128, lane, acc);
  const int half = lane >> 4, nlane = lane & 15;
#pragma unroll
  for (int nt = 0; nt < 8; ++nt) {
    int cn = nt * 16 + nlane;
#pragma unroll
    for (int g = 0; g < 8; ++g) {
      int r  = rowBase + half * 8 + g;
      int nd = nbase + r;
      if (nd < N) x1ws[(size_t)nd * 128 + cn] = (_Float16)acc[nt][g];
    }
  }
}

// ---------------------------------------------------------------------------
// GN2 edge MLP: in 368 (pad 384) -> 128 relu -> 64 ; atomic agg2 only
// ---------------------------------------------------------------------------
__global__ __launch_bounds__(64) void k_edge2(
    const float* __restrict__ x, const float* __restrict__ eattr,
    const _Float16* __restrict__ x1ws, const _Float16* __restrict__ e1ws,
    const int* __restrict__ row, const int* __restrict__ col,
    const _Float16* __restrict__ w1p, const float* __restrict__ b1,
    const _Float16* __restrict__ w2p, const float* __restrict__ b2,
    float* __restrict__ agg2, int E) {
  __shared__ alignas(16) _Float16 Ain[32 * 384];
  __shared__ alignas(16) _Float16 Hs[32 * 128];
  __shared__ int rowL[32], colL[32];
  const int tid   = threadIdx.x;
  const int ebase = blockIdx.x * 32;
  if (tid < 32) {
    int e = ebase + tid;
    rowL[tid] = (e < E) ? row[e] : 0;
    colL[tid] = (e < E) ? col[e] : 0;
  }
  __syncthreads();
  // 48 chunks/row: layout f = [x_r(2) x1_r(16) x_c(2) x1_c(16) ea(2) e1(8) pad(2)]
  for (int t = tid; t < 32 * 48; t += 64) {
    int r = t / 48, c = t % 48;
    int e = ebase + r;
    int ec = (e < E) ? e : (E - 1);
    int rs = rowL[r], cs = colL[r];
    _Float16* dst = &Ain[r * 384 + c * 8];
    if (c < 2)       *(v8h*)dst = cvt8(x + (size_t)rs * 16 + c * 8);
    else if (c < 18) cp16(dst, x1ws + (size_t)rs * 128 + (c - 2) * 8);
    else if (c < 20) *(v8h*)dst = cvt8(x + (size_t)cs * 16 + (c - 18) * 8);
    else if (c < 36) cp16(dst, x1ws + (size_t)cs * 128 + (c - 20) * 8);
    else if (c < 38) *(v8h*)dst = cvt8(eattr + (size_t)ec * 16 + (c - 36) * 8);
    else if (c < 46) cp16(dst, e1ws + (size_t)ec * 64 + (c - 38) * 8);
    else             *(v8h*)dst = (v8h){};
  }
  async_fence();
  __syncthreads();
  const int lane = tid & 31, wave = tid >> 5;
  const int rowBase = wave * 16;
  gemm1_relu<12, 8, 384>(Ain, rowBase, w1p, b1, Hs, lane);
  __syncthreads();
  v8f acc[4];
  gemm2<4>(Hs, rowBase, w2p, b2, 64, lane, acc);
  const int half = lane >> 4, nlane = lane & 15;
#pragma unroll
  for (int nt = 0; nt < 4; ++nt) {
    int cn = nt * 16 + nlane;
#pragma unroll
    for (int g = 0; g < 8; ++g) {
      int r = rowBase + half * 8 + g;
      int e = ebase + r;
      if (e < E) atomicAdd(&agg2[(size_t)colL[r] * 64 + cn], acc[nt][g]);
    }
  }
}

// ---------------------------------------------------------------------------
// GN2 node MLP: in 208 (pad 224) -> 128 relu -> 4 (pad 16) ; store final out
// ---------------------------------------------------------------------------
__global__ __launch_bounds__(128) void k_node2(
    const float* __restrict__ x, const _Float16* __restrict__ x1ws,
    const float* __restrict__ agg2,
    const _Float16* __restrict__ w1p, const float* __restrict__ b1,
    const _Float16* __restrict__ w2p, const float* __restrict__ b2,
    float* __restrict__ out, int N) {
  __shared__ alignas(16) _Float16 Ain[64 * 224];
  __shared__ alignas(16) _Float16 Hs[64 * 128];
  const int tid   = threadIdx.x;
  const int nbase = blockIdx.x * 64;
  // 28 chunks/row: 2 from x, 16 from x1ws (f16), 8 from agg2, 2 zero pad.
  for (int t = tid; t < 64 * 28; t += 128) {
    int r = t / 28, c = t % 28;
    int nd = nbase + r;
    int nc = (nd < N) ? nd : (N - 1);
    _Float16* dst = &Ain[r * 224 + c * 8];
    if (c < 2)       *(v8h*)dst = cvt8(x + (size_t)nc * 16 + c * 8);
    else if (c < 18) cp16(dst, x1ws + (size_t)nc * 128 + (c - 2) * 8);
    else if (c < 26) *(v8h*)dst = cvt8(agg2 + (size_t)nc * 64 + (c - 18) * 8);
    else             *(v8h*)dst = (v8h){};
  }
  async_fence();
  __syncthreads();
  const int lane = tid & 31, wave = tid >> 5;
  const int rowBase = wave * 16;
  gemm1_relu<7, 8, 224>(Ain, rowBase, w1p, b1, Hs, lane);
  __syncthreads();
  v8f acc[1];
  gemm2<1>(Hs, rowBase, w2p, b2, 4, lane, acc);
  const int half = lane >> 4, nlane = lane & 15;
  if (nlane < 4) {
#pragma unroll
    for (int g = 0; g < 8; ++g) {
      int r  = rowBase + half * 8 + g;
      int nd = nbase + r;
      if (nd < N) out[(size_t)nd * 4 + nlane] = acc[0][g];
    }
  }
}

// ---------------------------------------------------------------------------
extern "C" void kernel_launch(void* const* d_in, const int* in_sizes, int n_in,
                              void* d_out, int out_size, void* d_ws,
                              size_t ws_size, hipStream_t stream) {
  const float* x     = (const float*)d_in[0];
  const float* eattr = (const float*)d_in[1];
  const float* e1w1  = (const float*)d_in[2];
  const float* e1b1  = (const float*)d_in[3];
  const float* e1w2  = (const float*)d_in[4];
  const float* e1b2  = (const float*)d_in[5];
  const float* n1w1  = (const float*)d_in[6];
  const float* n1b1  = (const float*)d_in[7];
  const float* n1w2  = (const float*)d_in[8];
  const float* n1b2  = (const float*)d_in[9];
  const float* e2w1  = (const float*)d_in[10];
  const float* e2b1  = (const float*)d_in[11];
  const float* e2w2  = (const float*)d_in[12];
  const float* e2b2  = (const float*)d_in[13];
  const float* n2w1  = (const float*)d_in[14];
  const float* n2b1  = (const float*)d_in[15];
  const float* n2w2  = (const float*)d_in[16];
  const float* n2b2  = (const float*)d_in[17];
  const int*   eidx  = (const int*)d_in[18];

  const int E = in_sizes[18] / 2;
  const int N = in_sizes[0] / 16;
  const int* row = eidx;
  const int* col = eidx + E;

  char* p = (char*)d_ws;
  auto carve = [&](size_t bytes) -> char* {
    char* r = p;
    p += (bytes + 255) & ~(size_t)255;
    return r;
  };

  _Float16* e1ws = (_Float16*)carve((size_t)E * 64 * 2);
  _Float16* x1ws = (_Float16*)carve((size_t)N * 128 * 2);
  float*    agg1 = (float*)carve((size_t)N * 64 * 4 * 2);  // agg1 + agg2
  float*    agg2 = agg1 + (size_t)N * 64;

  auto packBytes = [](int KT, int NT) { return (size_t)KT * NT * 512 * 2; };
  _Float16* p_e1w1 = (_Float16*)carve(packBytes(2, 8));
  _Float16* p_e1w2 = (_Float16*)carve(packBytes(4, 4));
  _Float16* p_n1w1 = (_Float16*)carve(packBytes(3, 8));
  _Float16* p_n1w2 = (_Float16*)carve(packBytes(4, 8));
  _Float16* p_e2w1 = (_Float16*)carve(packBytes(12, 8));
  _Float16* p_e2w2 = (_Float16*)carve(packBytes(4, 4));
  _Float16* p_n2w1 = (_Float16*)carve(packBytes(7, 8));
  _Float16* p_n2w2 = (_Float16*)carve(packBytes(4, 1));

  // Zero aggregation buffers (atomics accumulate into them).
  {
    long long nz = (long long)N * 64 * 2;
    k_zero<<<(int)((nz + 255) / 256), 256, 0, stream>>>(agg1, nz);
  }

  // Repack all 8 weight matrices into WMMA B-fragment layout (f16).
  auto repack = [&](const float* W, _Float16* o, int K, int Nn, int KT, int NT) {
    int nfrag = KT * NT;
    int nthr  = nfrag * 32;
    k_repack<<<(nthr + 63) / 64, 64, 0, stream>>>(W, o, K, Nn, NT, nfrag);
  };
  repack(e1w1, p_e1w1,  48, 128,  2, 8);
  repack(e1w2, p_e1w2, 128,  64,  4, 4);
  repack(n1w1, p_n1w1,  80, 128,  3, 8);
  repack(n1w2, p_n1w2, 128, 128,  4, 8);
  repack(e2w1, p_e2w1, 368, 128, 12, 8);
  repack(e2w2, p_e2w2, 128,  64,  4, 4);
  repack(n2w1, p_n2w1, 208, 128,  7, 8);
  repack(n2w2, p_n2w2, 128,   4,  4, 1);

  const int gEdge1 = (E + 63) / 64;
  const int gNode  = (N + 63) / 64;
  const int gEdge2 = (E + 31) / 32;

  k_edge1<<<gEdge1, 128, 0, stream>>>(x, eattr, row, col, p_e1w1, e1b1, p_e1w2,
                                      e1b2, e1ws, agg1, E);
  k_node1<<<gNode, 128, 0, stream>>>(x, agg1, p_n1w1, n1b1, p_n1w2, n1b2, x1ws,
                                     N);
  k_edge2<<<gEdge2, 64, 0, stream>>>(x, eattr, x1ws, e1ws, row, col, p_e2w1,
                                     e2b1, p_e2w2, e2b2, agg2, E);
  k_node2<<<gNode, 128, 0, stream>>>(x, x1ws, agg2, p_n2w1, n2b1, p_n2w2, n2b2,
                                     (float*)d_out, N);
}
